// EfficientMoELayer_84817014161796
// MI455X (gfx1250) — matrix operand order repaired and checked by reading, not compiled, not run on hardware
//
#include <hip/hip_runtime.h>

#define D_MODEL 1024
#define D_FF    4096
#define N_EXP   8
#define T_TOK   4096
#define CAP     4096
#define TILE_M  64

typedef __attribute__((ext_vector_type(16))) __bf16 v16bf;
typedef __attribute__((ext_vector_type(8)))  float  v8f;

// ---------------- workspace layout (bytes) ----------------
#define WS_CNT   0
#define WS_OFF   64
#define WS_TOK   128
#define WS_GATE  (WS_TOK + N_EXP * CAP * 4)                       // 131200
#define WS_W1T   ((size_t)(((WS_GATE + N_EXP * CAP * 4) + 255) & ~255))
#define W1T_BYTES ((size_t)N_EXP * D_MODEL * D_FF * 2)
#define WS_W2T   (WS_W1T + W1T_BYTES)
#define W2T_BYTES ((size_t)N_EXP * D_FF * D_MODEL * 2)
#define WS_H     (WS_W2T + W2T_BYTES)
// h capacity rows: 2*T + N_EXP*TILE_M padding

// =============================================================
// Kernel 1: router — one wave32 per token
// =============================================================
__global__ __launch_bounds__(256)
void moe_router(const float* __restrict__ x, const float* __restrict__ Wr,
                int* __restrict__ cnt, int* __restrict__ tok_list,
                float* __restrict__ gate_list) {
    const int wave = threadIdx.x >> 5;
    const int lane = threadIdx.x & 31;
    const int t = blockIdx.x * 8 + wave;

    float part[N_EXP];
#pragma unroll
    for (int e = 0; e < N_EXP; ++e) part[e] = 0.f;

    const float* xt = x + (size_t)t * D_MODEL;
    for (int d = lane; d < D_MODEL; d += 32) {
        const float xv = xt[d];
        const float* wr = Wr + (size_t)d * N_EXP;
#pragma unroll
        for (int e = 0; e < N_EXP; ++e) part[e] += xv * wr[e];
    }
#pragma unroll
    for (int e = 0; e < N_EXP; ++e) {
#pragma unroll
        for (int m = 16; m > 0; m >>= 1)
            part[e] += __shfl_xor(part[e], m, 32);
    }
    float mx = part[0];
#pragma unroll
    for (int e = 1; e < N_EXP; ++e) mx = fmaxf(mx, part[e]);
    float p[N_EXP];
#pragma unroll
    for (int e = 0; e < N_EXP; ++e) p[e] = __expf(part[e] - mx);

    int i1 = 0; float v1 = p[0];
#pragma unroll
    for (int e = 1; e < N_EXP; ++e) if (p[e] > v1) { v1 = p[e]; i1 = e; }
    int i2 = (i1 == 0) ? 1 : 0; float v2 = p[i2];
#pragma unroll
    for (int e = 0; e < N_EXP; ++e)
        if (e != i1 && e != i2 && p[e] > v2) { v2 = p[e]; i2 = e; }

    const float inv = 1.f / (v1 + v2);
    if (lane == 0) {
        int s1 = atomicAdd(&cnt[i1], 1);
        tok_list[i1 * CAP + s1]  = t;
        gate_list[i1 * CAP + s1] = v1 * inv;
        int s2 = atomicAdd(&cnt[i2], 1);
        tok_list[i2 * CAP + s2]  = t;
        gate_list[i2 * CAP + s2] = v2 * inv;
    }
}

// =============================================================
// Kernel 2: TILE_M-aligned exclusive scan of per-expert counts
// =============================================================
__global__ void moe_scan(const int* __restrict__ cnt, int* __restrict__ off) {
    if (threadIdx.x == 0) {
        int a = 0;
#pragma unroll
        for (int e = 0; e < N_EXP; ++e) { off[e] = a; a += (cnt[e] + TILE_M - 1) & ~(TILE_M - 1); }
        off[N_EXP] = a;
    }
}

// =============================================================
// Kernel 3: fp32 [E][K][N] -> bf16 [E][N][K] transpose+convert
// grid: (N/32, K/32, E), block 256
// =============================================================
__global__ __launch_bounds__(256)
void cvt_transpose(const float* __restrict__ src, __bf16* __restrict__ dst,
                   int K, int N) {
    const int e  = blockIdx.z;
    const int n0 = blockIdx.x * 32;
    const int k0 = blockIdx.y * 32;
    const float* s = src + (size_t)e * K * N;
    __bf16*      d = dst + (size_t)e * N * K;

    __shared__ float tile[32][33];
    const int tx  = threadIdx.x & 31;
    const int ty0 = threadIdx.x >> 5;          // 0..7
#pragma unroll
    for (int i = 0; i < 4; ++i) {
        const int ky = ty0 + i * 8;
        tile[ky][tx] = s[(size_t)(k0 + ky) * N + (n0 + tx)];
    }
    __syncthreads();
    const int c2 = (threadIdx.x & 15) * 2;     // k pair within tile
    const int r0 = threadIdx.x >> 4;           // 0..15
#pragma unroll
    for (int i = 0; i < 2; ++i) {
        const int r = r0 + i * 16;             // n within tile
        __bf16 pk[2] = { (__bf16)tile[c2][r], (__bf16)tile[c2 + 1][r] };
        *(unsigned int*)&d[(size_t)(n0 + r) * K + (k0 + c2)] = *(unsigned int*)pk;
    }
}

// ---------- fragment loaders (layouts per CDNA5 ISA) ----------
// B 32x16: lane<16 -> N=lane, K={k0..k0+7, k0+16..k0+23}; lane>=16 -> N=lane-16, K +8
__device__ __forceinline__ v16bf load_bfrag(const __bf16* __restrict__ Bt, int ldK,
                                            int k0, int n0, int lane) {
    const __bf16* p = Bt + (size_t)(n0 + (lane & 15)) * ldK + k0 + ((lane >> 4) << 3);
    v16bf b;
    ((uint4*)&b)[0] = *(const uint4*)p;
    ((uint4*)&b)[1] = *(const uint4*)(p + 16);
    return b;
}
// A 16x32: lane<16 -> M=lane, K={k0..k0+7, k0+16..k0+23}; lane>=16 -> +8
template <typename PT>
__device__ __forceinline__ v16bf load_afrag(const PT* rowBase, int k0, int lane) {
    const int ka = k0 + ((lane >> 4) << 3);
    v16bf a;
    ((uint4*)&a)[0] = *(const uint4*)(rowBase + ka);
    ((uint4*)&a)[1] = *(const uint4*)(rowBase + ka + 16);
    return a;
}

// =============================================================
// Kernel 4: h = relu(x_gathered @ W1t[e]^T + b1[e])
// grid: (D_FF/256, CAP/TILE_M, N_EXP), block 256 (8 waves)
// wave: 2 M-tiles x 4 N-tiles = 8 WMMAs per k-step
// =============================================================
#define LDSTRIDE (D_MODEL + 8)

__global__ __launch_bounds__(256)
void moe_ffn1(const float* __restrict__ x, const __bf16* __restrict__ W1t,
              const float* __restrict__ b1,
              const int* __restrict__ cnt, const int* __restrict__ off,
              const int* __restrict__ tok_list, __bf16* __restrict__ h) {
    const int e = blockIdx.z;
    const int n_cnt = cnt[e];
    const int t0 = blockIdx.y * TILE_M;
    if (t0 >= n_cnt) return;

    __shared__ __bf16 sx[TILE_M * LDSTRIDE];

    const int tid = threadIdx.x;
    const int* tl = tok_list + e * CAP;

    // stage x tile (fp32 -> bf16) into LDS
    for (int q = tid; q < TILE_M * (D_MODEL / 4); q += 256) {
        const int row = q >> 8;
        const int c4  = q & 255;
        const int slot = t0 + row;
        const int tok = (slot < n_cnt) ? tl[slot] : tl[t0];
        const float4 v = ((const float4*)(x + (size_t)tok * D_MODEL))[c4];
        __bf16 pk[4] = { (__bf16)v.x, (__bf16)v.y, (__bf16)v.z, (__bf16)v.w };
        *(uint2*)&sx[row * LDSTRIDE + c4 * 4] = *(uint2*)pk;
    }
    __syncthreads();

    const int lane   = tid & 31;
    const int wv     = tid >> 5;
    const int mSel   = wv & 1;           // rows [mSel*32, +32)
    const int nGroup = wv >> 1;          // 64-wide column group
    const int nBase  = blockIdx.x * 256 + nGroup * 64;
    const __bf16* W1e = W1t + (size_t)e * D_MODEL * D_FF;

    v8f acc[2][4] = {};
    const __bf16* arow0 = &sx[(size_t)(mSel * 32 +      (lane & 15)) * LDSTRIDE];
    const __bf16* arow1 = &sx[(size_t)(mSel * 32 + 16 + (lane & 15)) * LDSTRIDE];

    for (int k0 = 0; k0 < D_MODEL; k0 += 32) {
        __builtin_prefetch(W1e + (size_t)nBase * D_MODEL + k0 + 64, 0, 1);
        const v16bf a0 = load_afrag(arow0, k0, lane);
        const v16bf a1 = load_afrag(arow1, k0, lane);
#pragma unroll
        for (int s = 0; s < 4; ++s) {
            const v16bf b = load_bfrag(W1e, D_MODEL, k0, nBase + s * 16, lane);
            acc[0][s] = __builtin_amdgcn_wmma_f32_16x16x32_bf16(
                false, a0, false, b, (short)0, acc[0][s], false, false);
            acc[1][s] = __builtin_amdgcn_wmma_f32_16x16x32_bf16(
                false, a1, false, b, (short)0, acc[1][s], false, false);
        }
    }

    const float* b1e = b1 + (size_t)e * D_FF;
    const int hBase = off[e] + t0;
#pragma unroll
    for (int s = 0; s < 4; ++s) {
        const int n = nBase + s * 16 + (lane & 15);
        const float bv = b1e[n];
#pragma unroll
        for (int mt = 0; mt < 2; ++mt) {
#pragma unroll
            for (int r = 0; r < 8; ++r) {
                const int m = mSel * 32 + mt * 16 + ((lane >> 4) << 3) + r;
                if (t0 + m < n_cnt) {
                    float vv = acc[mt][s][r] + bv;
                    vv = vv > 0.f ? vv : 0.f;
                    h[(size_t)(hBase + m) * D_FF + n] = (__bf16)vv;
                }
            }
        }
    }
}

// =============================================================
// Kernel 5: out += gate * (h @ W2t[e]^T + b2[e])
// grid: (D_MODEL/256, CAP/TILE_M, N_EXP), block 256
// =============================================================
__global__ __launch_bounds__(256)
void moe_ffn2(const __bf16* __restrict__ h, const __bf16* __restrict__ W2t,
              const float* __restrict__ b2,
              const int* __restrict__ cnt, const int* __restrict__ off,
              const int* __restrict__ tok_list, const float* __restrict__ gate_list,
              float* __restrict__ out) {
    const int e = blockIdx.z;
    const int n_cnt = cnt[e];
    const int t0 = blockIdx.y * TILE_M;
    if (t0 >= n_cnt) return;

    const int tid    = threadIdx.x;
    const int lane   = tid & 31;
    const int wv     = tid >> 5;
    const int mSel   = wv & 1;
    const int nGroup = wv >> 1;
    const int nBase  = blockIdx.x * 256 + nGroup * 64;
    const __bf16* W2e = W2t + (size_t)e * D_FF * D_MODEL;

    const __bf16* ha0 = h + (size_t)(off[e] + t0 + mSel * 32 +      (lane & 15)) * D_FF;
    const __bf16* ha1 = h + (size_t)(off[e] + t0 + mSel * 32 + 16 + (lane & 15)) * D_FF;

    v8f acc[2][4] = {};
    for (int k0 = 0; k0 < D_FF; k0 += 32) {
        __builtin_prefetch(ha0 + k0 + 64, 0, 1);
        const v16bf a0 = load_afrag(ha0, k0, lane);
        const v16bf a1 = load_afrag(ha1, k0, lane);
#pragma unroll
        for (int s = 0; s < 4; ++s) {
            const v16bf b = load_bfrag(W2e, D_FF, k0, nBase + s * 16, lane);
            acc[0][s] = __builtin_amdgcn_wmma_f32_16x16x32_bf16(
                false, a0, false, b, (short)0, acc[0][s], false, false);
            acc[1][s] = __builtin_amdgcn_wmma_f32_16x16x32_bf16(
                false, a1, false, b, (short)0, acc[1][s], false, false);
        }
    }

    const float* b2e = b2 + (size_t)e * D_MODEL;
    const int* tl = tok_list + e * CAP;
    const float* gl = gate_list + e * CAP;

    int   ncol[4];
    float bv[4];
#pragma unroll
    for (int s = 0; s < 4; ++s) {
        ncol[s] = nBase + s * 16 + (lane & 15);
        bv[s]   = b2e[ncol[s]];
    }
#pragma unroll
    for (int mt = 0; mt < 2; ++mt) {
#pragma unroll
        for (int r = 0; r < 8; ++r) {
            const int m = mSel * 32 + mt * 16 + ((lane >> 4) << 3) + r;
            const int slot = t0 + m;
            if (slot < n_cnt) {
                const int   tok = tl[slot];
                const float g   = gl[slot];
#pragma unroll
                for (int s = 0; s < 4; ++s)
                    atomicAdd(&out[(size_t)tok * D_MODEL + ncol[s]],
                              g * (acc[mt][s][r] + bv[s]));
            }
        }
    }
}

// =============================================================
extern "C" void kernel_launch(void* const* d_in, const int* in_sizes, int n_in,
                              void* d_out, int out_size, void* d_ws, size_t ws_size,
                              hipStream_t stream) {
    const float* x  = (const float*)d_in[0];
    const float* Wr = (const float*)d_in[1];
    const float* W1 = (const float*)d_in[2];
    const float* b1 = (const float*)d_in[3];
    const float* W2 = (const float*)d_in[4];
    const float* b2 = (const float*)d_in[5];
    float* out = (float*)d_out;

    char* ws = (char*)d_ws;
    int*    cnt  = (int*)(ws + WS_CNT);
    int*    off  = (int*)(ws + WS_OFF);
    int*    tok  = (int*)(ws + WS_TOK);
    float*  gate = (float*)(ws + WS_GATE);
    __bf16* w1t  = (__bf16*)(ws + WS_W1T);
    __bf16* w2t  = (__bf16*)(ws + WS_W2T);
    __bf16* h    = (__bf16*)(ws + WS_H);

    hipMemsetAsync(cnt, 0, N_EXP * sizeof(int), stream);
    hipMemsetAsync(out, 0, (size_t)out_size * sizeof(float), stream);

    moe_router<<<T_TOK / 8, 256, 0, stream>>>(x, Wr, cnt, tok, gate);
    moe_scan<<<1, 32, 0, stream>>>(cnt, off);

    // W1: [E][1024][4096] -> bf16 [E][4096][1024]
    cvt_transpose<<<dim3(D_FF / 32, D_MODEL / 32, N_EXP), 256, 0, stream>>>(
        W1, w1t, D_MODEL, D_FF);
    // W2: [E][4096][1024] -> bf16 [E][1024][4096]
    cvt_transpose<<<dim3(D_MODEL / 32, D_FF / 32, N_EXP), 256, 0, stream>>>(
        W2, w2t, D_FF, D_MODEL);

    dim3 g1(D_FF / 256, CAP / TILE_M, N_EXP);
    moe_ffn1<<<g1, 256, 0, stream>>>(x, w1t, b1, cnt, off, tok, h);

    dim3 g2(D_MODEL / 256, CAP / TILE_M, N_EXP);
    moe_ffn2<<<g2, 256, 0, stream>>>(h, w2t, b2, cnt, off, tok, gate, out);
}
